// EmbeddingBlock_57707180589097
// MI455X (gfx1250) — matrix operand order repaired and checked by reading, not compiled, not run on hardware
//
#include <hip/hip_runtime.h>
#include <hip/hip_bf16.h>

// ---------- types ----------
typedef __attribute__((ext_vector_type(16))) __bf16 v16bf;
typedef __attribute__((ext_vector_type(8)))  float  v8f;

union Frag16 {
    v16bf v;
    unsigned short u[16];
    uint4 q[2];
};

static __device__ __forceinline__ unsigned short f2bf(float f) {
    unsigned int u = __float_as_uint(f);
    unsigned int r = (u + 0x7FFFu + ((u >> 16) & 1u)) >> 16;
    return (unsigned short)r;
}

static __device__ __forceinline__ float silu_f(float x) {
    return x / (1.0f + __expf(-x));
}

#define FEAT 28
#define HH   64     // H/2
#define H    128
#define RDIM 6
#define BN_EPS 1e-5f

// ---------- kernel 0: zero stats ----------
__global__ void k_init_stats(float* stats) {
    if (threadIdx.x < 128) stats[threadIdx.x] = 0.0f;   // sum[64] + sumsq[64]
}

// ---------- kernel 1: h1 = x@W1 + b1, accumulate BN stats ----------
__global__ void k_node1(const float* __restrict__ x, const float* __restrict__ w1,
                        const float* __restrict__ b1, float* __restrict__ h1,
                        float* __restrict__ gSum, float* __restrict__ gSsq, int N) {
    __shared__ float sW1[FEAT * HH];   // 7 KB
    __shared__ float sB1[HH];
    __shared__ float sSum[HH];
    __shared__ float sSsq[HH];
    __shared__ float sX[8][FEAT];

    for (int idx = threadIdx.x; idx < FEAT * HH; idx += 256) sW1[idx] = w1[idx];
    if (threadIdx.x < HH) {
        sB1[threadIdx.x] = b1[threadIdx.x];
        sSum[threadIdx.x] = 0.0f;
        sSsq[threadIdx.x] = 0.0f;
    }
    __syncthreads();

    int wave = threadIdx.x >> 5;
    int lane = threadIdx.x & 31;
    int node = blockIdx.x * 8 + wave;

    if (node < N && lane < FEAT) sX[wave][lane] = x[(long)node * FEAT + lane];
    __syncthreads();

    if (node < N) {
#pragma unroll
        for (int g = 0; g < 2; g++) {
            int o = lane + 32 * g;
            float s = sB1[o];
#pragma unroll
            for (int t = 0; t < FEAT; t++) s += sX[wave][t] * sW1[t * HH + o];
            h1[(long)node * HH + o] = s;
            atomicAdd(&sSum[o], s);
            atomicAdd(&sSsq[o], s * s);
        }
    }
    __syncthreads();
    if (threadIdx.x < HH) {
        atomicAdd(&gSum[threadIdx.x], sSum[threadIdx.x]);
        atomicAdd(&gSsq[threadIdx.x], sSsq[threadIdx.x]);
    }
}

// ---------- kernel 2: finalize BN scale/shift ----------
__global__ void k_stats(const float* __restrict__ gSum, const float* __restrict__ gSsq,
                        const float* __restrict__ gamma, const float* __restrict__ beta,
                        float* __restrict__ scale, float* __restrict__ shift, int N) {
    int c = threadIdx.x;
    if (c < HH) {
        float inv = 1.0f / (float)N;
        float mean = gSum[c] * inv;
        float var  = gSsq[c] * inv - mean * mean;
        float sc   = gamma[c] * rsqrtf(var + BN_EPS);
        scale[c] = sc;
        shift[c] = beta[c] - mean * sc;
    }
}

// ---------- kernel 3: h2 = relu(relu(bn(h1)) @ W2 + b2), stored bf16 ----------
__global__ void k_node2(const float* __restrict__ h1, const float* __restrict__ w2,
                        const float* __restrict__ b2, const float* __restrict__ scale,
                        const float* __restrict__ shift, unsigned short* __restrict__ h2,
                        int N) {
    __shared__ float sW2[HH * H];   // 32 KB
    __shared__ float sB2[H];
    __shared__ float sScale[HH];
    __shared__ float sShift[HH];
    __shared__ float sU[8][HH];

    for (int idx = threadIdx.x; idx < HH * H; idx += 256) sW2[idx] = w2[idx];
    if (threadIdx.x < H)  sB2[threadIdx.x] = b2[threadIdx.x];
    if (threadIdx.x < HH) { sScale[threadIdx.x] = scale[threadIdx.x];
                            sShift[threadIdx.x] = shift[threadIdx.x]; }
    __syncthreads();

    int wave = threadIdx.x >> 5;
    int lane = threadIdx.x & 31;
    int node = blockIdx.x * 8 + wave;

    if (node < N) {
#pragma unroll
        for (int g = 0; g < 2; g++) {
            int k = lane + 32 * g;
            float v = h1[(long)node * HH + k] * sScale[k] + sShift[k];
            sU[wave][k] = fmaxf(v, 0.0f);
        }
    }
    __syncthreads();

    if (node < N) {
        float acc[4] = {0.0f, 0.0f, 0.0f, 0.0f};
#pragma unroll
        for (int k = 0; k < HH; k++) {
            float u = sU[wave][k];
#pragma unroll
            for (int c = 0; c < 4; c++)
                acc[c] += u * sW2[k * H + lane + 32 * c];
        }
#pragma unroll
        for (int c = 0; c < 4; c++) {
            int o = lane + 32 * c;
            float v = fmaxf(acc[c] + sB2[o], 0.0f);
            h2[(long)node * H + o] = f2bf(v);
        }
    }
}

// ---------- kernel 4: fused edge GEMM (WMMA bf16) ----------
// cat = [h[i], h[j], silu(rbf@rbf_w+rbf_b)]  -> silu(cat @ lin_w + lin_b)
// Each block handles 64 of the 128 output columns (halfsel), 8 waves,
// one 16-edge tile per wave per iteration.
__global__ void k_edge(const unsigned short* __restrict__ h2,
                       const int* __restrict__ idx_i, const int* __restrict__ idx_j,
                       const float* __restrict__ rbf,
                       const float* __restrict__ rbf_w, const float* __restrict__ rbf_b,
                       const float* __restrict__ lin_w, const float* __restrict__ lin_b,
                       float* __restrict__ out, int E, int nTiles) {
    // B fragments pre-swizzled: entry (hc, nt, n) = 16 bf16 of column n, chunk K 8*(hc&1)+0..7 then 16+8*(hc&1)+0..7
    __shared__ uint4 sWq[24 * 4 * 16 * 2];   // 48 KB
    __shared__ float sRbfW[RDIM * H];        // 3 KB
    __shared__ float sRbfB[H];
    __shared__ float sLinB[64];

    const int halfsel = blockIdx.x & 1;
    const int pairId  = blockIdx.x >> 1;
    const int nPairs  = gridDim.x >> 1;

    // stage lin_w half (384 x 64) as bf16 in WMMA-B per-lane layout
    unsigned short* sWu = (unsigned short*)sWq;
    for (int idx = threadIdx.x; idx < 384 * 64; idx += 256) {
        int k = idx >> 6;
        int n = idx & 63;
        float w = lin_w[k * H + halfsel * 64 + n];
        int hc = ((k >> 5) << 1) | ((k >> 3) & 1), kk = (k & 7) | ((k >> 1) & 8), nt = n >> 4, nn = n & 15;
        int ent = (hc * 4 + nt) * 16 + nn;
        sWu[ent * 16 + kk] = f2bf(w);
    }
    for (int idx = threadIdx.x; idx < RDIM * H; idx += 256) sRbfW[idx] = rbf_w[idx];
    if (threadIdx.x < H)  sRbfB[threadIdx.x] = rbf_b[threadIdx.x];
    if (threadIdx.x < 64) sLinB[threadIdx.x] = lin_b[halfsel * 64 + threadIdx.x];
    __syncthreads();

    const int wave = threadIdx.x >> 5;
    const int lane = threadIdx.x & 31;
    const int m    = lane & 15;
    const int kh   = lane >> 4;     // half-of-wave (A: K+8*kh ; B: K+8*kh)

    for (int tile = pairId * 8 + wave; tile < nTiles; tile += nPairs * 8) {
        // Compiler barrier: prevents LICM from hoisting the per-chunk LDS
        // B-fragment loads out of the tile loop (which spilled to scratch).
        asm volatile("" ::: "memory");

        long e  = (long)tile * 16 + m;
        long ec = (e < E) ? e : (long)(E - 1);     // clamp; tails masked at store
        int ri = idx_i[ec];
        int rj = idx_j[ec];
        float rb[RDIM];
#pragma unroll
        for (int t = 0; t < RDIM; t++) rb[t] = rbf[ec * RDIM + t];

        v8f acc[4];
#pragma unroll
        for (int nt = 0; nt < 4; nt++)
#pragma unroll
            for (int q = 0; q < 8; q++) acc[nt][q] = 0.0f;

        // ---- chunks 0..7 : A gathered from h2 rows (bf16, L2-resident) ----
#pragma unroll
        for (int c = 0; c < 8; c++) {
            Frag16 a;
            int row = (c < 4) ? ri : rj;
            const unsigned short* rowp = h2 + (long)row * H + (c & 3) * 32 + kh * 8;
            a.q[0] = *(const uint4*)(rowp);
            a.q[1] = *(const uint4*)(rowp + 16);

            Frag16 b[4];
#pragma unroll
            for (int nt = 0; nt < 4; nt++) {
                int ent = ((2 * c + kh) * 4 + nt) * 16 + m;
                b[nt].q[0] = sWq[ent * 2];
                b[nt].q[1] = sWq[ent * 2 + 1];
            }
#pragma unroll
            for (int nt = 0; nt < 4; nt++)
                acc[nt] = __builtin_amdgcn_wmma_f32_16x16x32_bf16(
                    false, a.v, false, b[nt].v, (short)0, acc[nt], false, false);
        }

        // ---- chunks 8..11 : A = silu(rbf @ rbf_w + rbf_b), computed in-frag ----
#pragma unroll
        for (int c = 8; c < 12; c++) {
            Frag16 a;
            int fbase = (c - 8) * 32 + kh * 8;
#pragma unroll
            for (int g = 0; g < 2; g++) {
#pragma unroll
                for (int kk = 0; kk < 8; kk++) {
                    int f = fbase + g * 16 + kk;
                    float s = sRbfB[f];
#pragma unroll
                    for (int t = 0; t < RDIM; t++) s += rb[t] * sRbfW[t * H + f];
                    a.u[g * 8 + kk] = f2bf(silu_f(s));
                }
            }

            Frag16 b[4];
#pragma unroll
            for (int nt = 0; nt < 4; nt++) {
                int ent = ((2 * c + kh) * 4 + nt) * 16 + m;
                b[nt].q[0] = sWq[ent * 2];
                b[nt].q[1] = sWq[ent * 2 + 1];
            }
#pragma unroll
            for (int nt = 0; nt < 4; nt++)
                acc[nt] = __builtin_amdgcn_wmma_f32_16x16x32_bf16(
                    false, a.v, false, b[nt].v, (short)0, acc[nt], false, false);
        }

        // D layout: n = lane&15 ; M = v + 8*(lane>=16)
#pragma unroll
        for (int nt = 0; nt < 4; nt++) {
            float bias = sLinB[nt * 16 + m];
#pragma unroll
            for (int v = 0; v < 8; v++) {
                int  mo = v + 8 * kh;
                long eo = (long)tile * 16 + mo;
                if (eo < E) {
                    float s = acc[nt][v] + bias;
                    out[eo * H + halfsel * 64 + nt * 16 + m] = silu_f(s);
                }
            }
        }
    }
}

// ---------- launch ----------
extern "C" void kernel_launch(void* const* d_in, const int* in_sizes, int n_in,
                              void* d_out, int out_size, void* d_ws, size_t ws_size,
                              hipStream_t stream) {
    const float* x      = (const float*)d_in[0];
    const float* rbf    = (const float*)d_in[1];
    const int*   iidx   = (const int*)d_in[2];
    const int*   jidx   = (const int*)d_in[3];
    const float* emb_w1 = (const float*)d_in[4];
    const float* emb_b1 = (const float*)d_in[5];
    const float* gamma  = (const float*)d_in[6];
    const float* beta   = (const float*)d_in[7];
    const float* emb_w2 = (const float*)d_in[8];
    const float* emb_b2 = (const float*)d_in[9];
    const float* rbf_w  = (const float*)d_in[10];
    const float* rbf_b  = (const float*)d_in[11];
    const float* lin_w  = (const float*)d_in[12];
    const float* lin_b  = (const float*)d_in[13];
    float* out = (float*)d_out;

    const int N = in_sizes[0] / FEAT;
    const int E = in_sizes[2];

    // workspace layout
    float* stats  = (float*)d_ws;          // sum[64] | sumsq[64] | scale[64] | shift[64]
    float* gSum   = stats;
    float* gSsq   = stats + 64;
    float* scale  = stats + 128;
    float* shift  = stats + 192;
    float* h1     = stats + 256;                         // N*64 f32
    unsigned short* h2 = (unsigned short*)(h1 + (size_t)N * HH);  // N*128 bf16

    k_init_stats<<<1, 128, 0, stream>>>(stats);

    int nodeBlocks = (N + 7) / 8;
    k_node1<<<nodeBlocks, 256, 0, stream>>>(x, emb_w1, emb_b1, h1, gSum, gSsq, N);
    k_stats<<<1, 64, 0, stream>>>(gSum, gSsq, gamma, beta, scale, shift, N);
    k_node2<<<nodeBlocks, 256, 0, stream>>>(h1, emb_w2, emb_b2, scale, shift, h2, N);

    int nTiles = (E + 15) / 16;
    k_edge<<<1024, 256, 0, stream>>>(h2, iidx, jidx, rbf, rbf_w, rbf_b,
                                     lin_w, lin_b, out, E, nTiles);
}
